// StdModel_82729660056160
// MI455X (gfx1250) — compile-verified, matrix operand order checked by reading, not run on hardware
//
#include <hip/hip_runtime.h>
#include <hip/hip_bf16.h>

// ---------------- model dims ----------------
#define VSZ   32000
#define DM    1024
#define NHEAD 16
#define FFD   4096
#define NLAY  2
#define HDIM  64
#define BB    2
#define TT    2048
#define MTOK  (BB*TT)          // 4096 tokens
#define EPSV  1.1920929e-07f

typedef __bf16 bf16;
typedef __attribute__((ext_vector_type(16))) __bf16 v16bf;
typedef __attribute__((ext_vector_type(8)))  __bf16 v8bf;
typedef __attribute__((ext_vector_type(4)))  __bf16 v4bf;
typedef __attribute__((ext_vector_type(8)))  float  v8f;
typedef __attribute__((ext_vector_type(4)))  float  v4f;

__device__ inline bf16 f2bf(float f) {
    unsigned u = __builtin_bit_cast(unsigned, f);
    unsigned r = u + 0x7fffu + ((u >> 16) & 1u);   // round-to-nearest-even
    return __builtin_bit_cast(bf16, (unsigned short)(r >> 16));
}

// Per-lane 16-bf16 WMMA fragment (A layout; B symmetric):
// lanes 0-15 hold K {0..7,16..23}, lanes 16-31 hold K {8..15,24..31}
// -> two 16-byte loads at p and p+16 (ds_load_b128).
__device__ inline v16bf load_frag(const bf16* __restrict__ p) {
    v8bf lo = *(const v8bf*)(p);
    v8bf hi = *(const v8bf*)(p + 16);
    return __builtin_shufflevector(lo, hi,
               0,1,2,3,4,5,6,7,8,9,10,11,12,13,14,15);
}

// ---- CDNA5 async copy: global -> LDS, 16B per lane, tracked by ASYNCcnt ----
__device__ inline void async_load_b128(const bf16* g, bf16* l) {
    // generic LDS pointer keeps the LDS byte offset in addr[31:0]
    unsigned lds = (unsigned)(uintptr_t)l;
    asm volatile("global_load_async_to_lds_b128 %0, %1, off"
                 :: "v"(lds), "v"(g) : "memory");
}
template<int N>
__device__ inline void wait_asynccnt() {
    asm volatile("s_wait_asynccnt %0" :: "i"(N) : "memory");
}

// =====================================================================
// WMMA GEMM:  C[M,N] = alpha*(A @ W^T) + bias (+ C if ACCUM)
//   A : bf16 [M,K] row-major,  W : bf16 [N,K] row-major,  K % 64 == 0
// Block = 256 threads = 8 waves (4(M) x 2(N)); wave tile 32x32 (2x2 WMMA);
// block tile 128(M) x 64(N).  Grid = (N/64, M/128).
// K-slices of 64 (128x64 A, 64x64 W) double-buffered in LDS via
// global_load_async_to_lds_b128 (6 async ops / wave / step, wait<=6);
// main loop peeled so the steady state is branch-free:
//   issue(next) -> s_wait_asynccnt 6 -> barrier -> 8x wmma -> barrier.
// =====================================================================
#define KSTEP      64
#define LDS_STRIDE 72   // 64 cols + 8 pad: 144B rows, 16B aligned, conflict-free

template<int ACCUM>
__global__ __launch_bounds__(256) void gemm_wmma_kernel(
    const bf16* __restrict__ A, const bf16* __restrict__ W,
    const float* __restrict__ bias, float* __restrict__ C,
    int M, int N, int K, int ldc, float alpha)
{
    __shared__ bf16 Asm[2][128 * LDS_STRIDE];   // 36 KB
    __shared__ bf16 Wsm[2][64 * LDS_STRIDE];    // 18 KB

    const int tid  = threadIdx.x;
    const int wave = tid >> 5;
    const int lane = tid & 31;
    const int half = lane >> 4;
    const int r    = lane & 15;

    const int blockM = blockIdx.y * 128;
    const int blockN = blockIdx.x * 64;

    // ---- async staging map: 1536 16B chunks / K-step, 6 per thread ----
    // A tile = 128 rows x 8 chunks; thread covers rows arow+{0,32,64,96}
    // W tile =  64 rows x 8 chunks; thread covers rows wrow+{0,32}
    const int arow = tid >> 3, acol = (tid & 7) * 8;   // col in elems
    const int al0 = (arow +  0) * LDS_STRIDE + acol;
    const int al1 = (arow + 32) * LDS_STRIDE + acol;
    const int al2 = (arow + 64) * LDS_STRIDE + acol;
    const int al3 = (arow + 96) * LDS_STRIDE + acol;
    const int wl0 = (arow +  0) * LDS_STRIDE + acol;
    const int wl1 = (arow + 32) * LDS_STRIDE + acol;

    const bf16* Ag0 = A + (size_t)(blockM + arow +  0) * K + acol;
    const bf16* Ag1 = A + (size_t)(blockM + arow + 32) * K + acol;
    const bf16* Ag2 = A + (size_t)(blockM + arow + 64) * K + acol;
    const bf16* Ag3 = A + (size_t)(blockM + arow + 96) * K + acol;
    const bf16* Wg0 = W + (size_t)(blockN + arow +  0) * K + acol;
    const bf16* Wg1 = W + (size_t)(blockN + arow + 32) * K + acol;

    const int mw = (wave >> 1) * 32;      // wave offsets inside block tile
    const int nw = (wave & 1)  * 32;

    v8f acc00 = {}, acc01 = {}, acc10 = {}, acc11 = {};

    auto issue = [&](int buf) {
        bf16* as = &Asm[buf][0];
        bf16* wsp = &Wsm[buf][0];
        async_load_b128(Ag0, as + al0);
        async_load_b128(Ag1, as + al1);
        async_load_b128(Ag2, as + al2);
        async_load_b128(Ag3, as + al3);
        async_load_b128(Wg0, wsp + wl0);
        async_load_b128(Wg1, wsp + wl1);
        // prefetch one K-step ahead of the async engine (global_prefetch_b8)
        __builtin_prefetch(Ag0 + KSTEP, 0, 1);
        __builtin_prefetch(Ag1 + KSTEP, 0, 1);
        __builtin_prefetch(Ag2 + KSTEP, 0, 1);
        __builtin_prefetch(Ag3 + KSTEP, 0, 1);
        __builtin_prefetch(Wg0 + KSTEP, 0, 1);
        __builtin_prefetch(Wg1 + KSTEP, 0, 1);
        Ag0 += KSTEP; Ag1 += KSTEP; Ag2 += KSTEP; Ag3 += KSTEP;
        Wg0 += KSTEP; Wg1 += KSTEP;
    };

    auto compute = [&](int buf) {
        const bf16* As = &Asm[buf][0];
        const bf16* Ws = &Wsm[buf][0];
        #pragma unroll
        for (int ks = 0; ks < KSTEP; ks += 32) {
            const int c = ks + half * 8;
            v16bf a0 = load_frag(As + (mw + r)      * LDS_STRIDE + c);
            v16bf a1 = load_frag(As + (mw + r + 16) * LDS_STRIDE + c);
            v16bf b0 = load_frag(Ws + (nw + r)      * LDS_STRIDE + c);
            v16bf b1 = load_frag(Ws + (nw + r + 16) * LDS_STRIDE + c);
            acc00 = __builtin_amdgcn_wmma_f32_16x16x32_bf16(
                        false, a0, false, b0, (short)0, acc00, false, false);
            acc01 = __builtin_amdgcn_wmma_f32_16x16x32_bf16(
                        false, a0, false, b1, (short)0, acc01, false, false);
            acc10 = __builtin_amdgcn_wmma_f32_16x16x32_bf16(
                        false, a1, false, b0, (short)0, acc10, false, false);
            acc11 = __builtin_amdgcn_wmma_f32_16x16x32_bf16(
                        false, a1, false, b1, (short)0, acc11, false, false);
        }
    };

    // prologue: stage K-slice 0 into buffer 0
    int buf = 0;
    issue(0);
    // steady state: branch-free issue/wait/compute with double buffering
    for (int kk = 0; kk + KSTEP < K; kk += KSTEP) {
        issue(buf ^ 1);
        wait_asynccnt<6>();       // previous slice complete (6 still in flight)
        __syncthreads();          // all waves' async writes visible
        compute(buf);
        __syncthreads();          // reads done before buffer reuse
        buf ^= 1;
    }
    // epilogue: final K-slice
    wait_asynccnt<0>();
    __syncthreads();
    compute(buf);

    const int mBase = blockM + mw, nBase = blockN + nw;
    const int col0 = nBase + r, col1 = col0 + 16;
    const float bv0 = bias ? bias[col0] : 0.0f;
    const float bv1 = bias ? bias[col1] : 0.0f;
    #pragma unroll
    for (int i = 0; i < 8; ++i) {
        const int row0 = mBase + half * 8 + i;   // C layout: VGPR i -> M=i / i+8
        const int row1 = row0 + 16;
        const size_t i00 = (size_t)row0 * ldc + col0;
        const size_t i01 = (size_t)row0 * ldc + col1;
        const size_t i10 = (size_t)row1 * ldc + col0;
        const size_t i11 = (size_t)row1 * ldc + col1;
        float v00 = alpha * acc00[i] + bv0;
        float v01 = alpha * acc01[i] + bv1;
        float v10 = alpha * acc10[i] + bv0;
        float v11 = alpha * acc11[i] + bv1;
        if (ACCUM) { v00 += C[i00]; v01 += C[i01]; v10 += C[i10]; v11 += C[i11]; }
        C[i00] = v00; C[i01] = v01; C[i10] = v10; C[i11] = v11;
    }
}

// ---------------- embedding gather ----------------
__global__ __launch_bounds__(256) void embed_kernel(
    const int* __restrict__ x, const float* __restrict__ emb,
    float* __restrict__ h)
{
    int i = blockIdx.x * 256 + threadIdx.x;
    if (i >= MTOK * DM) return;
    int tok = i / DM, d = i % DM;
    h[i] = emb[(size_t)x[tok] * DM + d];
}

// ---------------- RMSNorm -> bf16 ----------------
__global__ __launch_bounds__(256) void rmsnorm_kernel(
    const float* __restrict__ x, const float* __restrict__ w,
    bf16* __restrict__ out, int ncols)
{
    const int row = blockIdx.x;
    const float* xr = x + (size_t)row * ncols;
    __shared__ float red[256];
    float s = 0.f;
    for (int c = threadIdx.x; c < ncols; c += 256) { float v = xr[c]; s += v * v; }
    red[threadIdx.x] = s; __syncthreads();
    for (int off = 128; off > 0; off >>= 1) {
        if (threadIdx.x < off) red[threadIdx.x] += red[threadIdx.x + off];
        __syncthreads();
    }
    const float scale = rsqrtf(red[0] / ncols + EPSV);
    bf16* orow = out + (size_t)row * ncols;
    for (int c = threadIdx.x; c < ncols; c += 256)
        orow[c] = f2bf(xr[c] * scale * w[c]);
}

// ---------------- split qkv: Q,K bf16 [B,NH,T,HD]; V^T bf16 [B,NH,HD,T] ----------------
__global__ __launch_bounds__(256) void split_qkv_kernel(
    const float* __restrict__ qkv, bf16* __restrict__ Q,
    bf16* __restrict__ Kc, bf16* __restrict__ Vt)
{
    int i = blockIdx.x * 256 + threadIdx.x;   // over MTOK*DM
    if (i >= MTOK * DM) return;
    const int d = i & (HDIM - 1);
    const int h = (i >> 6) & (NHEAD - 1);
    const int t = (i / DM) & (TT - 1);
    const int b = i / (TT * DM);
    const float* base = qkv + (size_t)(b * TT + t) * (3 * DM) + h * HDIM + d;
    const size_t bh = (size_t)(b * NHEAD + h);
    Q [bh * TT * HDIM + (size_t)t * HDIM + d] = f2bf(base[0]);
    Kc[bh * TT * HDIM + (size_t)t * HDIM + d] = f2bf(base[DM]);
    Vt[bh * HDIM * TT + (size_t)d * TT + t]   = f2bf(base[2 * DM]);
}

// ---------------- causal softmax row -> bf16 probabilities ----------------
__global__ __launch_bounds__(256) void softmax_causal_kernel(
    const float* __restrict__ S, bf16* __restrict__ P, int T)
{
    const int q = blockIdx.x;
    const float* srow = S + (size_t)q * T;
    bf16* prow = P + (size_t)q * T;
    const int n = q + 1;                     // keys <= q are valid
    __shared__ float red[256];

    float m = -3.4e38f;
    for (int c = threadIdx.x; c < n; c += 256) m = fmaxf(m, srow[c]);
    red[threadIdx.x] = m; __syncthreads();
    for (int off = 128; off > 0; off >>= 1) {
        if (threadIdx.x < off)
            red[threadIdx.x] = fmaxf(red[threadIdx.x], red[threadIdx.x + off]);
        __syncthreads();
    }
    m = red[0]; __syncthreads();

    float s = 0.f;
    for (int c = threadIdx.x; c < n; c += 256) s += __expf(srow[c] - m);
    red[threadIdx.x] = s; __syncthreads();
    for (int off = 128; off > 0; off >>= 1) {
        if (threadIdx.x < off) red[threadIdx.x] += red[threadIdx.x + off];
        __syncthreads();
    }
    const float inv = 1.0f / red[0];

    for (int c = threadIdx.x; c < T; c += 256)
        prow[c] = (c < n) ? f2bf(__expf(srow[c] - m) * inv) : f2bf(0.0f);
}

// ---------------- f32 -> bf16, 4 elements per thread ----------------
__global__ __launch_bounds__(256) void cvt4_bf16_kernel(
    const float* __restrict__ in, bf16* __restrict__ out, int n4)
{
    int i = blockIdx.x * 256 + threadIdx.x;
    if (i >= n4) return;
    v4f v = ((const v4f*)in)[i];
    v4bf o; o[0] = f2bf(v[0]); o[1] = f2bf(v[1]); o[2] = f2bf(v[2]); o[3] = f2bf(v[3]);
    ((v4bf*)out)[i] = o;
}

__global__ __launch_bounds__(256) void swiglu_kernel(
    const float* __restrict__ g, const float* __restrict__ u,
    bf16* __restrict__ out, int n)
{
    int i = blockIdx.x * 256 + threadIdx.x;
    if (i < n) {
        float gv = g[i];
        float sil = gv / (1.0f + __expf(-gv));
        out[i] = f2bf(sil * u[i]);
    }
}

// =====================================================================
// Orchestration
// =====================================================================
extern "C" void kernel_launch(void* const* d_in, const int* in_sizes, int n_in,
                              void* d_out, int out_size, void* d_ws, size_t ws_size,
                              hipStream_t stream) {
    (void)in_sizes; (void)n_in; (void)out_size; (void)ws_size;
    const int*   x      = (const int*)  d_in[0];
    const float* emb_w  = (const float*)d_in[1];
    const float* n1_w   = (const float*)d_in[2];
    const float* n2_w   = (const float*)d_in[3];
    const float* qkv_w  = (const float*)d_in[4];
    const float* qkv_b  = (const float*)d_in[5];
    const float* o_w    = (const float*)d_in[6];
    const float* o_b    = (const float*)d_in[7];
    const float* g_w    = (const float*)d_in[8];
    const float* g_b    = (const float*)d_in[9];
    const float* u_w    = (const float*)d_in[10];
    const float* u_b    = (const float*)d_in[11];
    const float* dn_w   = (const float*)d_in[12];
    const float* dn_b   = (const float*)d_in[13];
    const float* norm_w = (const float*)d_in[14];
    const float* head_w = (const float*)d_in[15];
    float* out = (float*)d_out;

    // ---- workspace carve-up (~250 MB) ----
    char* ws = (char*)d_ws;
    size_t off = 0;
    auto alloc = [&](size_t bytes) -> void* {
        void* p = ws + off;
        off += (bytes + 255) & ~(size_t)255;
        return p;
    };
    float* h   = (float*)alloc((size_t)MTOK * DM * 4);
    bf16*  nb  = (bf16*) alloc((size_t)MTOK * DM * 2);
    float* qkv = (float*)alloc((size_t)MTOK * 3 * DM * 4);
    bf16*  Q   = (bf16*) alloc((size_t)MTOK * DM * 2);
    bf16*  Kc  = (bf16*) alloc((size_t)MTOK * DM * 2);
    bf16*  Vt  = (bf16*) alloc((size_t)MTOK * DM * 2);
    float* S   = (float*)alloc((size_t)TT * TT * 4);
    bf16*  P   = (bf16*) alloc((size_t)TT * TT * 2);
    float* ao  = (float*)alloc((size_t)MTOK * DM * 4);
    bf16*  ab  = (bf16*) alloc((size_t)MTOK * DM * 2);
    float* gch = (float*)alloc((size_t)1024 * FFD * 4);
    float* uch = (float*)alloc((size_t)1024 * FFD * 4);
    bf16*  fch = (bf16*) alloc((size_t)1024 * FFD * 2);
    // bf16 weight staging region: per-layer weights (16.8M elems) or head (32.77M)
    bf16*  wrg = (bf16*) alloc((size_t)VSZ * DM * 2);
    bf16*  wqkv = wrg;
    bf16*  wo   = wrg + (size_t)3 * DM * DM;
    bf16*  wg   = wo  + (size_t)DM * DM;
    bf16*  wu   = wg  + (size_t)FFD * DM;
    bf16*  wdn  = wu  + (size_t)FFD * DM;

    const dim3 blk(256);
    auto gemm = [&](const bf16* A, const bf16* W, const float* bias, float* C,
                    int M, int N, int K, int ldc, float alpha, int accum) {
        dim3 grid(N / 64, M / 128);
        if (accum)
            gemm_wmma_kernel<1><<<grid, blk, 0, stream>>>(A, W, bias, C, M, N, K, ldc, alpha);
        else
            gemm_wmma_kernel<0><<<grid, blk, 0, stream>>>(A, W, bias, C, M, N, K, ldc, alpha);
    };
    auto cvtw = [&](const float* src, bf16* dst, size_t n) {
        int n4 = (int)(n / 4);
        cvt4_bf16_kernel<<<(n4 + 255) / 256, blk, 0, stream>>>(src, dst, n4);
    };

    const int elemsMD = MTOK * DM;
    embed_kernel<<<(elemsMD + 255) / 256, blk, 0, stream>>>(x, emb_w, h);

    for (int l = 0; l < NLAY; ++l) {
        // ---- stage this layer's weights as bf16 ----
        cvtw(qkv_w + (size_t)l * 3 * DM * DM, wqkv, (size_t)3 * DM * DM);
        cvtw(o_w   + (size_t)l * DM * DM,     wo,   (size_t)DM * DM);
        cvtw(g_w   + (size_t)l * FFD * DM,    wg,   (size_t)FFD * DM);
        cvtw(u_w   + (size_t)l * FFD * DM,    wu,   (size_t)FFD * DM);
        cvtw(dn_w  + (size_t)l * DM * FFD,    wdn,  (size_t)DM * FFD);

        // ---- attention block ----
        rmsnorm_kernel<<<MTOK, blk, 0, stream>>>(h, n1_w + l * DM, nb, DM);
        gemm(nb, wqkv, qkv_b + (size_t)l * 3 * DM,
             qkv, MTOK, 3 * DM, DM, 3 * DM, 1.0f, 0);
        split_qkv_kernel<<<(elemsMD + 255) / 256, blk, 0, stream>>>(qkv, Q, Kc, Vt);

        for (int bh = 0; bh < BB * NHEAD; ++bh) {
            const bf16* Qbh = Q  + (size_t)bh * TT * HDIM;
            const bf16* Kbh = Kc + (size_t)bh * TT * HDIM;
            const bf16* Vbh = Vt + (size_t)bh * HDIM * TT;
            // S = (Q K^T) / sqrt(HD)
            gemm(Qbh, Kbh, nullptr, S, TT, TT, HDIM, TT, 0.125f, 0);
            softmax_causal_kernel<<<TT, blk, 0, stream>>>(S, P, TT);
            // O = P V  (V pre-transposed so it is A @ W^T with W = V^T)
            const int b = bh / NHEAD, hh = bh % NHEAD;
            float* Obh = ao + (size_t)b * TT * DM + (size_t)hh * HDIM;
            gemm(P, Vbh, nullptr, Obh, TT, HDIM, TT, DM, 1.0f, 0);
        }
        cvt4_bf16_kernel<<<(elemsMD / 4 + 255) / 256, blk, 0, stream>>>(ao, ab, elemsMD / 4);
        gemm(ab, wo, o_b + (size_t)l * DM, h, MTOK, DM, DM, DM, 1.0f, 1);   // h +=

        // ---- SwiGLU FFN (chunked over 1024-row slices) ----
        rmsnorm_kernel<<<MTOK, blk, 0, stream>>>(h, n2_w + l * DM, nb, DM);
        for (int mc = 0; mc < MTOK / 1024; ++mc) {
            const bf16* Anb = nb + (size_t)mc * 1024 * DM;
            gemm(Anb, wg, g_b + (size_t)l * FFD, gch, 1024, FFD, DM, FFD, 1.0f, 0);
            gemm(Anb, wu, u_b + (size_t)l * FFD, uch, 1024, FFD, DM, FFD, 1.0f, 0);
            swiglu_kernel<<<(1024 * FFD + 255) / 256, blk, 0, stream>>>(
                gch, uch, fch, 1024 * FFD);
            gemm(fch, wdn, dn_b + (size_t)l * DM,
                 h + (size_t)mc * 1024 * DM, 1024, DM, FFD, DM, 1.0f, 1);   // h +=
        }
    }

    // ---- final norm + LM head ----
    rmsnorm_kernel<<<MTOK, blk, 0, stream>>>(h, norm_w, nb, DM);
    cvtw(head_w, wrg, (size_t)VSZ * DM);
    gemm(nb, wrg, nullptr, out, MTOK, VSZ, DM, VSZ, 1.0f, 0);
}
